// S4DTransitionModel_30305289240680
// MI455X (gfx1250) — compile-verified, matrix operand order checked by reading, not run on hardware
//
#include <hip/hip_runtime.h>
#include <hip/hip_bf16.h>

// ---------------- problem constants ----------------
#define BATCH  8192
#define LATENT 256
#define CDIM   128
#define UDIM   32
#define NY     50
#define SELIN  289   // LATENT + UDIM + 1
#define KPAD   320   // SELIN padded up to multiple of 32
#define HZLD   264   // LDS row pitch for hz tile (256 + 8 halfs -> conflict-free)

// ---------------- workspace layout (f16 element offsets) ----------------
constexpr size_t S_W0  = 256 * (size_t)KPAD;     // selector w0 padded [256,320]
constexpr size_t S_W1  = 256 * 256;
constexpr size_t S_W2  = 256 * 256;
constexpr size_t S_AL  = 256 * 256;              // [alpha_w ; omega_w] stacked [256,256]
constexpr size_t S_BT  = 4096 * 256;             // btr / bti each
constexpr size_t S_CT  = 6400 * 256;             // ctr / cti each
constexpr size_t S_DT  = 1600 * 256;
constexpr size_t S_X   = (size_t)BATCH * KPAD;   // f16 input matrix
constexpr size_t S_ACT = (size_t)BATCH * 256;    // f16 activations

constexpr size_t O_W0 = 0;
constexpr size_t O_W1 = O_W0 + S_W0;
constexpr size_t O_W2 = O_W1 + S_W1;
constexpr size_t O_AL = O_W2 + S_W2;
constexpr size_t O_BTR = O_AL + S_AL;
constexpr size_t O_BTI = O_BTR + S_BT;
constexpr size_t O_CTR = O_BTI + S_BT;
constexpr size_t O_CTI = O_CTR + S_CT;
constexpr size_t O_DTW = O_CTI + S_CT;
constexpr size_t O_X   = O_DTW + S_DT;
constexpr size_t O_END = O_X + S_X;              // 8,683,520 f16 elems

// ---------------- WMMA helpers ----------------
typedef _Float16 v16h __attribute__((ext_vector_type(16)));
typedef float    v8f  __attribute__((ext_vector_type(8)));

struct __align__(16) F16x8 { _Float16 h[8]; };

// Load a 16x32 f16 fragment (A or B operand) from a row-major [rows, ld]
// matrix in GLOBAL memory: lane (&15) selects the row within the 16-row tile,
// elements hold K per the CDNA5 16-bit operand layout:
//   lanes 0-15 : elems 0..7 -> K k0+0..7,  elems 8..15 -> K k0+16..23
//   lanes 16-31: elems 0..7 -> K k0+8..15, elems 8..15 -> K k0+24..31
__device__ __forceinline__ v16h load_frag(const _Float16* __restrict__ base,
                                          int ld, int row_base, int k0) {
  int lane = threadIdx.x & 31;
  int grp  = lane >> 4;
  int idx  = lane & 15;
  const _Float16* p = base + (size_t)(row_base + idx) * (size_t)ld + k0 + grp * 8;
  F16x8 lo = *(const F16x8*)(p);
  F16x8 hi = *(const F16x8*)(p + 16);
  v16h f;
#pragma unroll
  for (int e = 0; e < 8; ++e) { f[e] = lo.h[e]; f[e + 8] = hi.h[e]; }
  return f;
}

// Same fragment pattern, but sourced from the LDS-staged hz tile
// (row pitch HZLD=264 halfs = 528B -> 16 lanes hit 64 distinct banks).
__device__ __forceinline__ v16h load_frag_hz(const _Float16 (&hzs)[32][HZLD],
                                             int row_base, int k0) {
  int lane = threadIdx.x & 31;
  int grp  = lane >> 4;
  int idx  = lane & 15;
  const _Float16* p = &hzs[row_base + idx][k0 + grp * 8];
  F16x8 lo = *(const F16x8*)(p);
  F16x8 hi = *(const F16x8*)(p + 16);
  v16h f;
#pragma unroll
  for (int e = 0; e < 8; ++e) { f[e] = lo.h[e]; f[e + 8] = hi.h[e]; }
  return f;
}

__device__ __forceinline__ v8f wmma_f16(v16h a, v16h b, v8f c) {
  return __builtin_amdgcn_wmma_f32_16x16x32_f16(false, a, false, b,
                                                (short)0, c, false, false);
}

// ---------------- kernel 1: weight f32->f16 conversion + input assembly ----------------
__global__ __launch_bounds__(256) void prep_kernel(
    const float* __restrict__ zt, const float* __restrict__ dt, const float* __restrict__ ut,
    const float* __restrict__ w0, const float* __restrict__ w1, const float* __restrict__ w2,
    const float* __restrict__ aw, const float* __restrict__ ow,
    const float* __restrict__ btrw, const float* __restrict__ btiw,
    const float* __restrict__ ctrw, const float* __restrict__ ctiw,
    const float* __restrict__ dtww,
    _Float16* __restrict__ wsh) {
  size_t i = (size_t)blockIdx.x * 256 + threadIdx.x;
  if (i >= O_END) return;
  float v;
  if (i < O_W1) {                       // selector w0, pad K 289 -> 320
    size_t r = i - O_W0; size_t n = r / KPAD, k = r % KPAD;
    v = (k < SELIN) ? w0[n * SELIN + k] : 0.f;
  } else if (i < O_W2) { v = w1[i - O_W1];
  } else if (i < O_AL) { v = w2[i - O_W2];
  } else if (i < O_BTR) {               // [alpha_w ; omega_w]
    size_t r = i - O_AL; size_t n = r >> 8, k = r & 255;
    v = (n < CDIM) ? aw[n * 256 + k] : ow[(n - CDIM) * 256 + k];
  } else if (i < O_BTI) { v = btrw[i - O_BTR];
  } else if (i < O_CTR) { v = btiw[i - O_BTI];
  } else if (i < O_CTI) { v = ctrw[i - O_CTR];
  } else if (i < O_DTW) { v = ctiw[i - O_CTI];
  } else if (i < O_X)   { v = dtww[i - O_DTW];
  } else {                              // x = [zt | ut | dt | 0]
    size_t r = i - O_X; size_t b = r / KPAD, k = r % KPAD;
    v = (k < 256)  ? zt[b * 256 + k]
      : (k < 288)  ? ut[b * UDIM + (k - 256)]
      : (k == 288) ? dt[b] : 0.f;
  }
  wsh[i] = (_Float16)v;
}

// ---------------- kernel 2: selector GEMM (M=8192, N=256) ----------------
// MODE 0: BN+ReLU epilogue -> f16 out
// MODE 1: +bias            -> f16 out
// MODE 2: +bias (split bias[0:128]/bias2[0:128]) -> f32 out (alpha/omega pre-acts)
template <int MODE>
__global__ __launch_bounds__(256) void sel_gemm(
    const _Float16* __restrict__ A, int kChunks, int lda,
    const _Float16* __restrict__ W,
    const float* __restrict__ bias, const float* __restrict__ bias2,
    const float* __restrict__ g, const float* __restrict__ be,
    _Float16* __restrict__ outH, float* __restrict__ outF) {
  int wave = threadIdx.x >> 5;
  int lane = threadIdx.x & 31;
  int grp = lane >> 4, idx = lane & 15;
  int mbase = blockIdx.x * 32 + (wave >> 2) * 16;
  int nbase = (wave & 3) * 64;

  v8f acc[4] = {};
  for (int kk = 0; kk < kChunks; ++kk) {
    v16h a = load_frag(A, lda, mbase, kk * 32);
#pragma unroll
    for (int t = 0; t < 4; ++t) {
      v16h b = load_frag(W, lda, nbase + t * 16, kk * 32);
      acc[t] = wmma_f16(a, b, acc[t]);
    }
  }
  const float inv = 0.9999950000374997f;  // 1/sqrt(1+1e-5)
#pragma unroll
  for (int t = 0; t < 4; ++t) {
#pragma unroll
    for (int j = 0; j < 8; ++j) {
      int m = mbase + j + grp * 8;
      int n = nbase + t * 16 + idx;
      float v = acc[t][j];
      if (MODE == 0) {
        v = fmaf((v + bias[n]) * inv, g[n], be[n]);
        v = fmaxf(v, 0.f);
        outH[(size_t)m * 256 + n] = (_Float16)v;
      } else if (MODE == 1) {
        outH[(size_t)m * 256 + n] = (_Float16)(v + bias[n]);
      } else {
        float bb = (n < CDIM) ? bias[n] : bias2[n - CDIM];
        outF[(size_t)m * 256 + n] = v + bb;
      }
    }
  }
}

// ---------------- kernel 3: fused heads (Bt/Ct/Dt GEMMs + contractions) ----------------
__global__ __launch_bounds__(256) void heads_kernel(
    const _Float16* __restrict__ hz,
    const _Float16* __restrict__ btrW, const _Float16* __restrict__ btiW,
    const _Float16* __restrict__ ctrW, const _Float16* __restrict__ ctiW,
    const _Float16* __restrict__ dtwW,
    const float* __restrict__ ao_pre,   // [B,256]: cols 0..127 alpha pre, 128..255 omega pre
    const float* __restrict__ zt, const float* __restrict__ dt, const float* __restrict__ ut,
    const float* __restrict__ btr_b, const float* __restrict__ bti_b,
    const float* __restrict__ ctr_b, const float* __restrict__ cti_b,
    const float* __restrict__ dtw_b,
    float* __restrict__ out) {
  __shared__ float zr[32][CDIM];
  __shared__ float zi[32][CDIM];
  __shared__ float Us[32][UDIM];
  __shared__ float yts[32][NY];
  __shared__ _Float16 hzs[32][HZLD];   // staged hz tile, padded pitch

  const int tid = threadIdx.x;
  const int wave = tid >> 5;
  const int lane = tid & 31;
  const int grp = lane >> 4, idx = lane & 15;
  const int bb = blockIdx.x * 32;

  // ---- init: stage hz tile to LDS, ut*dt, zero yt ----
  for (int p = tid; p < 32 * (LATENT / 8); p += 256) {   // 1024 8-half chunks
    int row = p >> 5, kc = (p & 31) * 8;
    *(F16x8*)(&hzs[row][kc]) = *(const F16x8*)(hz + (size_t)(bb + row) * 256 + kc);
  }
  for (int p = tid; p < 32 * UDIM; p += 256) {
    int m = p >> 5, u = p & 31;
    Us[m][u] = ut[(size_t)(bb + m) * UDIM + u] * dt[bb + m];
  }
  for (int p = tid; p < 32 * NY; p += 256) yts[p / NY][p % NY] = 0.f;
  __syncthreads();

  // ---- init z accumulators: Lam*z + Bt-bias dot ----
  for (int p = tid; p < 32 * CDIM; p += 256) {
    int m = p >> 7, c = p & 127;
    size_t b = bb + m;
    float ap = ao_pre[b * 256 + c];
    float om = ao_pre[b * 256 + 128 + c];
    float sp = fmaxf(ap, 0.f) + log1pf(expf(-fabsf(ap)));  // softplus
    float ea = expf(-sp);
    float lr = ea * cosf(om), li = ea * sinf(om);
    float z0r = zt[b * 256 + c], z0i = zt[b * 256 + 128 + c];
    float br = 0.f, bi = 0.f;
#pragma unroll 8
    for (int u = 0; u < UDIM; ++u) {
      float uv = Us[m][u];
      br = fmaf(btr_b[c * UDIM + u], uv, br);
      bi = fmaf(bti_b[c * UDIM + u], uv, bi);
    }
    zr[m][c] = lr * z0r - li * z0i + br;
    zi[m][c] = lr * z0i + li * z0r + bi;
  }
  __syncthreads();

  // ---- phase 1: Bt GEMM fused with sum_u Bt[b,c,u]*U[b,u] ----
  // items: mat(2:btr,bti) x mtile(2) x c(128)  -> wave-owned => race-free LDS adds
  for (int it = wave; it < 512; it += 8) {
    int mat = it >> 8, rem = it & 255, mt = rem >> 7, c = rem & 127;
    const _Float16* W = mat ? btiW : btrW;
    int n0 = c * UDIM;
    v8f acc0 = {}, acc1 = {};
    for (int kk = 0; kk < 8; ++kk) {
      v16h a  = load_frag_hz(hzs, mt * 16, kk * 32);
      v16h b0 = load_frag(W, 256, n0, kk * 32);
      v16h b1 = load_frag(W, 256, n0 + 16, kk * 32);
      acc0 = wmma_f16(a, b0, acc0);
      acc1 = wmma_f16(a, b1, acc1);
    }
#pragma unroll
    for (int j = 0; j < 8; ++j) {
      int row = mt * 16 + j + grp * 8;
      float s = acc0[j] * Us[row][idx] + acc1[j] * Us[row][16 + idx];
      s += __shfl_xor(s, 1); s += __shfl_xor(s, 2);
      s += __shfl_xor(s, 4); s += __shfl_xor(s, 8);
      if (idx == 0) { if (mat) zi[row][c] += s; else zr[row][c] += s; }
    }
  }
  __syncthreads();

  // ---- write z_next_real output [B,256] = [zr | zi] ----
  for (int p = tid; p < 32 * CDIM; p += 256) {
    int m = p >> 7, c = p & 127;
    size_t b = bb + m;
    out[b * 256 + c] = zr[m][c];
    out[b * 256 + 128 + c] = zi[m][c];
  }

  // ---- phase 2: Ct/Dt GEMM fused with contraction into yt ----
  // items: ctr 50y*8cc*2mt=800, cti 800, dtw 50y*2uc*2mt=200 -> 1800 (225/wave)
  for (int it = wave; it < 1800; it += 8) {
    const _Float16* W;
    int n0, y, cc, mt, kind;
    if (it < 800) {
      kind = 0; int e = it; y = e >> 4; int r = e & 15; cc = r >> 1; mt = r & 1;
      W = ctrW; n0 = y * CDIM + cc * 16;
    } else if (it < 1600) {
      kind = 1; int e = it - 800; y = e >> 4; int r = e & 15; cc = r >> 1; mt = r & 1;
      W = ctiW; n0 = y * CDIM + cc * 16;
    } else {
      kind = 2; int e = it - 1600; y = e >> 2; int r = e & 3; cc = r >> 1; mt = r & 1;
      W = dtwW; n0 = y * UDIM + cc * 16;
    }
    v8f acc = {};
    for (int kk = 0; kk < 8; ++kk) {
      v16h a = load_frag_hz(hzs, mt * 16, kk * 32);
      v16h b = load_frag(W, 256, n0, kk * 32);
      acc = wmma_f16(a, b, acc);
    }
#pragma unroll
    for (int j = 0; j < 8; ++j) {
      int row = mt * 16 + j + grp * 8;
      float mul = (kind == 0) ? zr[row][cc * 16 + idx]
                : (kind == 1) ? -zi[row][cc * 16 + idx]
                              : Us[row][cc * 16 + idx];
      float s = acc[j] * mul;
      s += __shfl_xor(s, 1); s += __shfl_xor(s, 2);
      s += __shfl_xor(s, 4); s += __shfl_xor(s, 8);
      if (idx == 0) atomicAdd(&yts[row][y], s);
    }
  }
  __syncthreads();

  // ---- phase 3: yt bias dots + store ----
  for (int p = tid; p < 32 * NY; p += 256) {
    int m = p / NY, y = p % NY;
    size_t b = bb + m;
    float acc = yts[m][y];
    for (int c = 0; c < CDIM; ++c)
      acc += ctr_b[y * CDIM + c] * zr[m][c] - cti_b[y * CDIM + c] * zi[m][c];
#pragma unroll 8
    for (int u = 0; u < UDIM; ++u)
      acc = fmaf(dtw_b[y * UDIM + u], Us[m][u], acc);
    out[(size_t)BATCH * 256 + b * NY + y] = acc;
  }
}

// ---------------- host launcher ----------------
extern "C" void kernel_launch(void* const* d_in, const int* in_sizes, int n_in,
                              void* d_out, int out_size, void* d_ws, size_t ws_size,
                              hipStream_t stream) {
  (void)in_sizes; (void)n_in; (void)out_size; (void)ws_size;
  const float* zt      = (const float*)d_in[0];
  const float* dt      = (const float*)d_in[1];
  const float* ut      = (const float*)d_in[2];
  const float* sel_w0  = (const float*)d_in[3];
  const float* sel_b0  = (const float*)d_in[4];
  const float* sel_g0  = (const float*)d_in[5];
  const float* sel_be0 = (const float*)d_in[6];
  const float* sel_w1  = (const float*)d_in[7];
  const float* sel_b1  = (const float*)d_in[8];
  const float* sel_g1  = (const float*)d_in[9];
  const float* sel_be1 = (const float*)d_in[10];
  const float* sel_w2  = (const float*)d_in[11];
  const float* sel_b2  = (const float*)d_in[12];
  const float* alpha_w = (const float*)d_in[13];
  const float* alpha_b = (const float*)d_in[14];
  const float* omega_w = (const float*)d_in[15];
  const float* omega_b = (const float*)d_in[16];
  const float* btr_w   = (const float*)d_in[17];
  const float* btr_b   = (const float*)d_in[18];
  const float* bti_w   = (const float*)d_in[19];
  const float* bti_b   = (const float*)d_in[20];
  const float* ctr_w   = (const float*)d_in[21];
  const float* ctr_b   = (const float*)d_in[22];
  const float* cti_w   = (const float*)d_in[23];
  const float* cti_b   = (const float*)d_in[24];
  const float* dtw_w   = (const float*)d_in[25];
  const float* dtw_b   = (const float*)d_in[26];

  _Float16* wsh = (_Float16*)d_ws;
  _Float16* xh  = wsh + O_X;
  _Float16* a0h = wsh + O_END;
  _Float16* a1h = a0h + S_ACT;
  _Float16* hzh = a1h + S_ACT;
  float*    ao  = (float*)(hzh + S_ACT);
  float*    out = (float*)d_out;

  // 1. weight conversion + input assembly
  prep_kernel<<<dim3((unsigned)((O_END + 255) / 256)), 256, 0, stream>>>(
      zt, dt, ut, sel_w0, sel_w1, sel_w2, alpha_w, omega_w,
      btr_w, bti_w, ctr_w, cti_w, dtw_w, wsh);

  // 2. selector MLP + alpha/omega pre-activations (all WMMA)
  dim3 gsel(BATCH / 32);
  sel_gemm<0><<<gsel, 256, 0, stream>>>(xh, KPAD / 32, KPAD, wsh + O_W0,
                                        sel_b0, nullptr, sel_g0, sel_be0, a0h, nullptr);
  sel_gemm<0><<<gsel, 256, 0, stream>>>(a0h, 8, 256, wsh + O_W1,
                                        sel_b1, nullptr, sel_g1, sel_be1, a1h, nullptr);
  sel_gemm<1><<<gsel, 256, 0, stream>>>(a1h, 8, 256, wsh + O_W2,
                                        sel_b2, nullptr, nullptr, nullptr, hzh, nullptr);
  sel_gemm<2><<<gsel, 256, 0, stream>>>(hzh, 8, 256, wsh + O_AL,
                                        alpha_b, omega_b, nullptr, nullptr, nullptr, ao);

  // 3. fused heads: Bt/Ct/Dt WMMA GEMMs + on-the-fly contractions
  heads_kernel<<<dim3(BATCH / 32), 256, 0, stream>>>(
      hzh, wsh + O_BTR, wsh + O_BTI, wsh + O_CTR, wsh + O_CTI, wsh + O_DTW,
      ao, zt, dt, ut, btr_b, bti_b, ctr_b, cti_b, dtw_b, out);
}
